// Attention_26388279067437
// MI455X (gfx1250) — matrix operand order repaired and checked by reading, not compiled, not run on hardware
//
#include <hip/hip_runtime.h>
#include <hip/hip_bf16.h>

// ---------------------------------------------------------------------------
// MHA forward for MI455X (gfx1250, wave32, WMMA).
//   D_MODEL=2048, HEADS=16, D_K=128, B=2, S=2048
// All heavy math on v_wmma_f32_16x16x32_bf16 (f32 accumulate).
// ---------------------------------------------------------------------------

constexpr int D_MODEL = 2048;
constexpr int NHEADS  = 16;
constexpr int D_K     = 128;
constexpr int BATCH   = 2;
constexpr int SEQ     = 2048;
constexpr int MTOK    = BATCH * SEQ;      // 4096 token rows

typedef __bf16 v16bf __attribute__((ext_vector_type(16)));
typedef float  v8f   __attribute__((ext_vector_type(8)));

union BFrag { v16bf v; unsigned int u[8]; };

__device__ __forceinline__ unsigned short f2bf(float f) {
    unsigned int x = __float_as_uint(f);
    return (unsigned short)((x + 0x7FFFu + ((x >> 16) & 1u)) >> 16);   // RNE
}
__device__ __forceinline__ unsigned int pk2(float a, float b) {
    return (unsigned int)f2bf(a) | ((unsigned int)f2bf(b) << 16);
}
__device__ __forceinline__ v8f wmma_bf16(const BFrag& a, const BFrag& b, v8f c) {
    return __builtin_amdgcn_wmma_f32_16x16x32_bf16(
        /*neg_a=*/false, a.v, /*neg_b=*/false, b.v,
        /*c_mod=*/(short)0, c, /*reuse_a=*/false, /*reuse_b=*/false);
}

// Output modes for the GEMM epilogue.
constexpr int OUT_F32        = 0;  // float [M,N]
constexpr int OUT_BF16       = 1;  // bf16  [M,N]
constexpr int OUT_BF16_TRANS = 2;  // bf16  Vt[b, h, d, s]  (per-head transpose)

// ---------------------------------------------------------------------------
// GEMM:  Y[M,N] = X[M,K] * W[N,K]^T + bias[N]     (nn.Linear convention)
// Block = 256 threads = 8 waves; tile 128x64, K-step 32.
// Wave grid 4x2: each wave owns a 32x32 patch -> 4 WMMAs/K-step,
// 2 A-frags x 2 B-frags fully reused.
// ---------------------------------------------------------------------------
template <int OMODE>
__global__ __launch_bounds__(256)
void gemm_bias(const float* __restrict__ X, const float* __restrict__ W,
               const float* __restrict__ bias, void* __restrict__ Yv,
               int M, int N, int K)
{
    __shared__ unsigned short Xs[128][34];  // bf16, +2 pad (row stride 68B, 4B-aligned)
    __shared__ unsigned short Ws[64][34];

    const int tid  = threadIdx.x;
    const int lane = tid & 31;
    const int wave = tid >> 5;
    const int half = lane >> 4;
    const int ln   = lane & 15;
    const int wm   = (wave >> 1) * 32;      // 0,32,64,96
    const int wn   = (wave & 1) * 32;       // 0,32
    const int bm   = blockIdx.y * 128;
    const int bn   = blockIdx.x * 64;

    // loader indices: X tile 128x32 (16 f32/thread), W tile 64x32 (8 f32/thread)
    const int xrow = tid >> 1;              // 0..127
    const int xcol = (tid & 1) * 16;        // 0,16
    const int wrow = tid >> 2;              // 0..63
    const int wcol = (tid & 3) * 8;         // 0,8,16,24

    v8f acc00 = {}, acc01 = {}, acc10 = {}, acc11 = {};

    for (int k0 = 0; k0 < K; k0 += 32) {
        __syncthreads();
        {   // stage X tile (f32 -> bf16)
            const float* src = X + (size_t)(bm + xrow) * K + k0 + xcol;
            float4 a = *(const float4*)src;
            float4 b = *(const float4*)(src + 4);
            float4 c = *(const float4*)(src + 8);
            float4 d = *(const float4*)(src + 12);
            unsigned int* dst = (unsigned int*)&Xs[xrow][xcol];
            dst[0] = pk2(a.x, a.y); dst[1] = pk2(a.z, a.w);
            dst[2] = pk2(b.x, b.y); dst[3] = pk2(b.z, b.w);
            dst[4] = pk2(c.x, c.y); dst[5] = pk2(c.z, c.w);
            dst[6] = pk2(d.x, d.y); dst[7] = pk2(d.z, d.w);
        }
        {   // stage W tile
            const float* src = W + (size_t)(bn + wrow) * K + k0 + wcol;
            float4 a = *(const float4*)src;
            float4 b = *(const float4*)(src + 4);
            unsigned int* dst = (unsigned int*)&Ws[wrow][wcol];
            dst[0] = pk2(a.x, a.y); dst[1] = pk2(a.z, a.w);
            dst[2] = pk2(b.x, b.y); dst[3] = pk2(b.z, b.w);
        }
        __syncthreads();

        BFrag b0, b1;                       // B fragments: lane = N-row, K pairs contiguous
#pragma unroll
        for (int v = 0; v < 8; ++v) {
            const int kk = half * 16 + 2 * v;
            b0.u[v] = *(const unsigned int*)&Ws[wn + ln][kk];
            b1.u[v] = *(const unsigned int*)&Ws[wn + 16 + ln][kk];
        }
        BFrag a0, a1;                       // A fragments: lane = M-row, K pairs contiguous
#pragma unroll
        for (int v = 0; v < 8; ++v) {
            const int kk = half * 16 + 2 * v;
            a0.u[v] = *(const unsigned int*)&Xs[wm + ln][kk];
            a1.u[v] = *(const unsigned int*)&Xs[wm + 16 + ln][kk];
        }
        acc00 = wmma_bf16(a0, b0, acc00);
        acc01 = wmma_bf16(a0, b1, acc01);
        acc10 = wmma_bf16(a1, b0, acc10);
        acc11 = wmma_bf16(a1, b1, acc11);
    }

    const int   gn0 = bn + wn + ln;
    const int   gn1 = gn0 + 16;
    const float bv0 = bias[gn0];
    const float bv1 = bias[gn1];

    if (OMODE == OUT_BF16_TRANS) {
        // Per-head transposed store: Vt[((b*NHEADS+h)*D_K + d) * SEQ + s].
        // 8 s-consecutive values per lane pack into one uint4 (b128) store.
        unsigned short* Vt = (unsigned short*)Yv;
        const int bb = bm >> 11;                    // SEQ = 2048, tile within batch
        const int sbase = (bm & (SEQ - 1)) + wm + half * 8;
        v8f accs[2][2] = {{acc00, acc01}, {acc10, acc11}};
#pragma unroll
        for (int mi = 0; mi < 2; ++mi) {
#pragma unroll
            for (int ni = 0; ni < 2; ++ni) {
                const int gn = ni ? gn1 : gn0;
                const float bvx = ni ? bv1 : bv0;
                const int hh = gn >> 7;             // D_K = 128
                const int dd = gn & (D_K - 1);
                v8f a = accs[mi][ni];
                uint4 p;
                p.x = pk2(a[0] + bvx, a[1] + bvx);
                p.y = pk2(a[2] + bvx, a[3] + bvx);
                p.z = pk2(a[4] + bvx, a[5] + bvx);
                p.w = pk2(a[6] + bvx, a[7] + bvx);
                unsigned short* dst = Vt +
                    ((size_t)(bb * NHEADS + hh) * D_K + dd) * SEQ + sbase + mi * 16;
                *(uint4*)dst = p;
            }
        }
    } else {
#pragma unroll
        for (int r = 0; r < 8; ++r) {               // C/D layout: m = half*8 + r, n = ln
            const int m0 = bm + wm + half * 8 + r;
            const int m1 = m0 + 16;
            const float v00 = acc00[r] + bv0, v01 = acc01[r] + bv1;
            const float v10 = acc10[r] + bv0, v11 = acc11[r] + bv1;
            if (OMODE == OUT_BF16) {
                unsigned short* Y = (unsigned short*)Yv;
                Y[(size_t)m0 * N + gn0] = f2bf(v00);
                Y[(size_t)m0 * N + gn1] = f2bf(v01);
                Y[(size_t)m1 * N + gn0] = f2bf(v10);
                Y[(size_t)m1 * N + gn1] = f2bf(v11);
            } else {
                float* Y = (float*)Yv;
                Y[(size_t)m0 * N + gn0] = v00;
                Y[(size_t)m0 * N + gn1] = v01;
                Y[(size_t)m1 * N + gn0] = v10;
                Y[(size_t)m1 * N + gn1] = v11;
            }
        }
    }
}

// ---------------------------------------------------------------------------
// Flash attention (causal, online softmax). One wave owns 16 query rows.
// Q/K bf16 [MTOK, D_MODEL] (head h at cols h*128..); V transposed per head:
// Vt[b, h, d, s] bf16 so V B-fragments are contiguous 32B per lane.
// Key tiles of 32; per tile: 8 WMMA for Q.K^T, 8 WMMA for P.V.
// No block barriers inside the loop (waves have different causal extents).
// ---------------------------------------------------------------------------
__global__ __launch_bounds__(256)
void attn_fwd(const unsigned short* __restrict__ Qb,
              const unsigned short* __restrict__ Kb,
              const unsigned short* __restrict__ Vt,
              float* __restrict__ O)
{
    __shared__ unsigned short sP[8][16][32];    // per-wave C->A layout bounce

    const int tid  = threadIdx.x;
    const int lane = tid & 31;
    const int wave = tid >> 5;
    const int half = lane >> 4;
    const int ln   = lane & 15;
    const int b    = blockIdx.z;
    const int h    = blockIdx.y;
    const int qbase = blockIdx.x * 128 + wave * 16;

    const size_t tok0 = (size_t)b * SEQ;
    const unsigned short* Qp  = Qb + (tok0 + qbase) * D_MODEL + h * D_K;
    const unsigned short* Kp  = Kb + tok0 * D_MODEL + h * D_K;
    const unsigned short* Vtp = Vt + (size_t)(b * NHEADS + h) * D_K * SEQ;

    // Load Q fragments once: 4 K-chunks of 32 along d_k.
    BFrag qf[4];
#pragma unroll
    for (int c = 0; c < 4; ++c)
#pragma unroll
        for (int v = 0; v < 8; ++v) {
            const int d = c * 32 + half * 16 + 2 * v;
            qf[c].u[v] = *(const unsigned int*)&Qp[(size_t)ln * D_MODEL + d];
        }

    float mi[8], li[8];
    v8f o[8] = {};
#pragma unroll
    for (int r = 0; r < 8; ++r) { mi[r] = -1e30f; li[r] = 0.0f; }

    const float sm_scale = 0.08838834764831845f;   // 1/sqrt(128)

    for (int j0 = 0; j0 < qbase + 16; j0 += 32) {
        v8f s0 = {}; v8f s1 = {};
#pragma unroll
        for (int c = 0; c < 4; ++c) {
            BFrag kf0, kf1;             // B fragments of K^T: lane = key row
#pragma unroll
            for (int v = 0; v < 8; ++v) {
                const int d = c * 32 + half * 16 + 2 * v;
                kf0.u[v] = *(const unsigned int*)&Kp[(size_t)(j0 + ln) * D_MODEL + d];
                kf1.u[v] = *(const unsigned int*)&Kp[(size_t)(j0 + 16 + ln) * D_MODEL + d];
            }
            s0 = wmma_bf16(qf[c], kf0, s0);
            s1 = wmma_bf16(qf[c], kf1, s1);
        }

        // scale + causal mask + row max (16-lane butterfly within half-group)
        float rowmax[8];
#pragma unroll
        for (int r = 0; r < 8; ++r) {
            const int qi = qbase + half * 8 + r;
            float a = s0[r] * sm_scale;
            float c = s1[r] * sm_scale;
            if (j0 + ln > qi)       a = -1e30f;
            if (j0 + 16 + ln > qi)  c = -1e30f;
            s0[r] = a; s1[r] = c;
            rowmax[r] = fmaxf(a, c);
        }
#pragma unroll
        for (int r = 0; r < 8; ++r) {
            float mx = rowmax[r];
#pragma unroll
            for (int off = 1; off < 16; off <<= 1)
                mx = fmaxf(mx, __shfl_xor(mx, off, 32));
            rowmax[r] = mx;
        }

        float pscale[8];
#pragma unroll
        for (int r = 0; r < 8; ++r) {
            const float mn = fmaxf(mi[r], rowmax[r]);
            pscale[r] = __expf(mi[r] - mn);
            const float p0 = __expf(s0[r] - mn);
            const float p1 = __expf(s1[r] - mn);
            s0[r] = p0; s1[r] = p1;
            float sum = p0 + p1;
#pragma unroll
            for (int off = 1; off < 16; off <<= 1)
                sum += __shfl_xor(sum, off, 32);
            li[r] = li[r] * pscale[r] + sum;
            mi[r] = mn;
        }
#pragma unroll
        for (int dt = 0; dt < 8; ++dt)
#pragma unroll
            for (int r = 0; r < 8; ++r)
                o[dt][r] *= pscale[r];

        // P: C layout -> A layout via per-wave LDS patch (same-wave DS order).
#pragma unroll
        for (int r = 0; r < 8; ++r) {
            sP[wave][half * 8 + r][ln]      = f2bf(s0[r]);
            sP[wave][half * 8 + r][16 + ln] = f2bf(s1[r]);
        }
        asm volatile("s_wait_dscnt 0" ::: "memory");

        BFrag pf;
#pragma unroll
        for (int v = 0; v < 8; ++v) {
            const int kk = half * 16 + 2 * v;
            pf.u[v] = *(const unsigned int*)&sP[wave][ln][kk];
        }

        // o += P (16x32) * V_tile (32x128): 8 N-subtiles of 16.
        // Vt is [d, s] per head -> per-lane 32B contiguous fragment loads.
#pragma unroll
        for (int dt = 0; dt < 8; ++dt) {
            BFrag vf;
            const unsigned short* vrow = Vtp + (size_t)(dt * 16 + ln) * SEQ + j0 + half * 16;
#pragma unroll
            for (int v = 0; v < 8; ++v)
                vf.u[v] = *(const unsigned int*)&vrow[2 * v];
            o[dt] = wmma_bf16(pf, vf, o[dt]);
        }
    }

    // epilogue: O[b, qbase+m, h*128 + d] = o / l
    float* Op = O + (tok0 + qbase) * D_MODEL + h * D_K;
#pragma unroll
    for (int r = 0; r < 8; ++r) {
        const float inv = 1.0f / li[r];
#pragma unroll
        for (int dt = 0; dt < 8; ++dt)
            Op[(size_t)(half * 8 + r) * D_MODEL + dt * 16 + ln] = o[dt][r] * inv;
    }
}

// ---------------------------------------------------------------------------
// Launch: QKV projections (bf16 out, V transposed per head) -> flash attention
// -> output projection. Workspace: Q,K,Vt bf16 (16 MB each) + O f32 (32 MB).
// ---------------------------------------------------------------------------
extern "C" void kernel_launch(void* const* d_in, const int* in_sizes, int n_in,
                              void* d_out, int out_size, void* d_ws, size_t ws_size,
                              hipStream_t stream)
{
    (void)in_sizes; (void)n_in; (void)out_size; (void)ws_size;
    const float* x  = (const float*)d_in[0];
    const float* wq = (const float*)d_in[1];
    const float* bq = (const float*)d_in[2];
    const float* wk = (const float*)d_in[3];
    const float* bk = (const float*)d_in[4];
    const float* wv = (const float*)d_in[5];
    const float* bv = (const float*)d_in[6];
    const float* wo = (const float*)d_in[7];
    const float* bo = (const float*)d_in[8];
    // d_in[9] = causal mask, implied by kernel structure.

    const size_t elems = (size_t)MTOK * D_MODEL;
    unsigned short* Qbf = (unsigned short*)d_ws;
    unsigned short* Kbf = Qbf + elems;
    unsigned short* Vtr = Kbf + elems;
    float*          Of  = (float*)(Vtr + elems);

    dim3 gGemm(D_MODEL / 64, MTOK / 128);          // (32, 32)
    dim3 bThr(256);

    gemm_bias<OUT_BF16><<<gGemm, bThr, 0, stream>>>(x, wq, bq, Qbf, MTOK, D_MODEL, D_MODEL);
    gemm_bias<OUT_BF16><<<gGemm, bThr, 0, stream>>>(x, wk, bk, Kbf, MTOK, D_MODEL, D_MODEL);
    gemm_bias<OUT_BF16_TRANS><<<gGemm, bThr, 0, stream>>>(x, wv, bv, Vtr, MTOK, D_MODEL, D_MODEL);

    dim3 gAttn(SEQ / 128, NHEADS, BATCH);          // (16, 16, 2)
    attn_fwd<<<gAttn, bThr, 0, stream>>>(Qbf, Kbf, Vtr, Of);

    gemm_bias<OUT_F32><<<gGemm, bThr, 0, stream>>>(Of, wo, bo, (float*)d_out,
                                                   MTOK, D_MODEL, D_MODEL);
}
